// SDGNNEncoder_26482768347666
// MI455X (gfx1250) — compile-verified, matrix operand order checked by reading, not run on hardware
//
#include <hip/hip_runtime.h>
#include <hip/hip_bf16.h>

typedef __attribute__((ext_vector_type(16))) _Float16 v16h;
typedef __attribute__((ext_vector_type(8)))  _Float16 v8h;
typedef __attribute__((ext_vector_type(8)))  float    v8f;

#define CDIV(a,b) (((a)+(b)-1)/(b))

// ---------- helpers: ordered-float encoding for integer atomicMax ----------
__device__ __forceinline__ unsigned f2ord(float f) {
    unsigned u = __float_as_uint(f);
    return (u & 0x80000000u) ? ~u : (u | 0x80000000u);
}
__device__ __forceinline__ float ord2f(unsigned o) {
    unsigned u = (o & 0x80000000u) ? (o & 0x7fffffffu) : ~o;
    return __uint_as_float(u);
}
__device__ __forceinline__ float lrelu(float e) { return e > 0.0f ? e : 0.2f * e; }

__device__ __forceinline__ v16h load_frag(const _Float16* row, int o0) {
    v8h lo = *(const v8h*)(row + o0);
    v8h hi = *(const v8h*)(row + o0 + 16);
    return __builtin_shufflevector(lo, hi, 0,1,2,3,4,5,6,7,8,9,10,11,12,13,14,15);
}

// ---------- WMMA GEMM (generic, tile per wave): C = A[rows,K] * Bt[cols,K]^T ----------
// Used for the K=576 MLP GEMM (A = comb_h stays L2-resident, re-reads are cheap).
__global__ void gemm_wmma_f16(const _Float16* __restrict__ A,
                              const _Float16* __restrict__ Bt,
                              float* __restrict__ Cmat,
                              int rowTiles, int colTiles, int K)
{
    int wid = blockIdx.x * (blockDim.x >> 5) + (threadIdx.x >> 5);
    if (wid >= rowTiles * colTiles) return;    // wave-uniform: EXEC stays all-1s
    int rt = wid % rowTiles;
    int ct = wid / rowTiles;
    int lane = threadIdx.x & 31;
    int lr = lane & 15;
    int hi = lane >> 4;

    const _Float16* arow = A  + (size_t)(rt * 16 + lr) * K;
    const _Float16* brow = Bt + (size_t)(ct * 16 + lr) * K;

    v8f c = {};
    for (int kw = 0; kw < K; kw += 32) {
        int o0 = kw + 8 * hi;   // CDNA5 16-bit A layout: lanes 0-15 K[0..7,16..23], lanes 16-31 K[8..15,24..31]
        v16h av = load_frag(arow, o0);
        v16h bv = load_frag(brow, o0);
        c = __builtin_amdgcn_wmma_f32_16x16x32_f16(false, av, false, bv, (short)0, c, false, false);
    }
    int ldc = colTiles * 16;
    float* crow = Cmat + (size_t)(rt * 16 + 8 * hi) * ldc + (size_t)(ct * 16 + lr);
    #pragma unroll
    for (int i = 0; i < 8; i++) crow[(size_t)i * ldc] = c[i];
}

// ---------- WMMA GEMM (row-block, K <= 64): wave holds A fragment in registers ----------
// and sweeps all column tiles -> A read exactly once from memory.
// OUT_F16 selects f16 epilogue (GAT features feeding the edge gather) vs f32.
template <bool OUT_F16>
__global__ void gemm_wmma_rowblock(const _Float16* __restrict__ A,
                                   const _Float16* __restrict__ Bt,
                                   void* __restrict__ Cout,
                                   int rowTiles, int colTiles, int K)
{
    int rt = blockIdx.x * (blockDim.x >> 5) + (threadIdx.x >> 5);
    if (rt >= rowTiles) return;
    int lane = threadIdx.x & 31;
    int lr = lane & 15;
    int hi = lane >> 4;
    int nk = K >> 5;                           // 1 or 2 k-windows (K <= 64)

    const _Float16* arow = A + (size_t)(rt * 16 + lr) * K;
    v16h af[2];
    for (int k = 0; k < nk; k++) af[k] = load_frag(arow, k * 32 + 8 * hi);

    int ldc = colTiles * 16;
    for (int ct = 0; ct < colTiles; ct++) {
        const _Float16* brow = Bt + (size_t)(ct * 16 + lr) * K;
        v8f c = {};
        for (int k = 0; k < nk; k++) {
            v16h bv = load_frag(brow, k * 32 + 8 * hi);
            c = __builtin_amdgcn_wmma_f32_16x16x32_f16(false, af[k], false, bv, (short)0, c,
                                                       false, false);
        }
        size_t base = (size_t)(rt * 16 + 8 * hi) * ldc + (size_t)(ct * 16 + lr);
        if (OUT_F16) {
            _Float16* crow = (_Float16*)Cout + base;
            #pragma unroll
            for (int i = 0; i < 8; i++) crow[(size_t)i * ldc] = (_Float16)c[i];
        } else {
            float* crow = (float*)Cout + base;
            #pragma unroll
            for (int i = 0; i < 8; i++) crow[(size_t)i * ldc] = c[i];
        }
    }
}

// ---------- conversions ----------
__global__ void cvt_f32_to_f16(const float* __restrict__ in, _Float16* __restrict__ out, int n)
{
    int i = blockIdx.x * blockDim.x + threadIdx.x;
    if (i < n) out[i] = (_Float16)in[i];
}

// in [batch][rows][cols] f32 -> out [batch][cols][rows] f16
__global__ void transpose_f32_f16(const float* __restrict__ in, _Float16* __restrict__ out,
                                  int rows, int cols, int batch)
{
    int i = blockIdx.x * blockDim.x + threadIdx.x;
    int tot = rows * cols * batch;
    if (i >= tot) return;
    int b = i / (rows * cols);
    int rem = i % (rows * cols);
    int r = rem / cols, c = rem % cols;
    out[(size_t)b * rows * cols + (size_t)c * rows + r] = (_Float16)in[i];
}

// ---------- GAT attention pieces (Hf is f16 [N][128], head-major cols) ----------
__global__ void esed_kernel(const _Float16* __restrict__ Hf,
                            const float* __restrict__ asrc, const float* __restrict__ adst,
                            float* __restrict__ es, float* __restrict__ ed, int n)
{
    int idx = blockIdx.x * blockDim.x + threadIdx.x;
    if (idx >= n * 2) return;
    int node = idx >> 1, head = idx & 1;
    const _Float16* hp = Hf + (size_t)node * 128 + head * 64;
    const float* ap = asrc + head * 64;
    const float* bp = adst + head * 64;
    float s = 0.f, d = 0.f;
    #pragma unroll 8
    for (int c = 0; c < 64; c++) { float hv = (float)hp[c]; s += hv * ap[c]; d += hv * bp[c]; }
    es[idx] = s; ed[idx] = d;
}

__global__ void attn_self_max(const float* __restrict__ es, const float* __restrict__ ed,
                              unsigned* __restrict__ emax_bits, int n2)
{
    int idx = blockIdx.x * blockDim.x + threadIdx.x;
    if (idx >= n2) return;
    emax_bits[idx] = f2ord(lrelu(es[idx] + ed[idx]));   // self-loop always present
}

__global__ void attn_edge_max(const int* __restrict__ srcA, const int* __restrict__ dstA,
                              const float* __restrict__ ew, int swap, int wantPos,
                              const float* __restrict__ es, const float* __restrict__ ed,
                              unsigned* __restrict__ emax_bits, int E)
{
    int e = blockIdx.x * blockDim.x + threadIdx.x;
    if (e >= E) return;
    float w = ew[e];
    bool m = wantPos ? (w > 0.0f) : (w < 0.0f);
    if (!m) return;
    int s = srcA[e], d = dstA[e];
    if (swap) { int t = s; s = d; d = t; }
    #pragma unroll
    for (int h = 0; h < 2; h++) {
        float v = lrelu(es[s * 2 + h] + ed[d * 2 + h]);
        atomicMax(&emax_bits[d * 2 + h], f2ord(v));
    }
}

__global__ void attn_den_init(const float* __restrict__ es, const float* __restrict__ ed,
                              const unsigned* __restrict__ emax_bits,
                              float* __restrict__ emaxf, float* __restrict__ pself,
                              float* __restrict__ den, int n2)
{
    int idx = blockIdx.x * blockDim.x + threadIdx.x;
    if (idx >= n2) return;
    float m = ord2f(emax_bits[idx]);
    float p = __expf(lrelu(es[idx] + ed[idx]) - m);
    emaxf[idx] = m; pself[idx] = p; den[idx] = p;
}

__global__ void attn_edge_p(const int* __restrict__ srcA, const int* __restrict__ dstA,
                            const float* __restrict__ ew, int swap, int wantPos,
                            const float* __restrict__ es, const float* __restrict__ ed,
                            const float* __restrict__ emaxf,
                            float* __restrict__ pbuf, float* __restrict__ den, int E)
{
    int e = blockIdx.x * blockDim.x + threadIdx.x;
    if (e >= E) return;
    float w = ew[e];
    bool m = wantPos ? (w > 0.0f) : (w < 0.0f);
    int s = srcA[e], d = dstA[e];
    if (swap) { int t = s; s = d; d = t; }
    #pragma unroll
    for (int h = 0; h < 2; h++) {
        float p = 0.f;
        if (m) {
            float v = lrelu(es[s * 2 + h] + ed[d * 2 + h]);
            p = __expf(v - emaxf[d * 2 + h]);
            atomicAdd(&den[d * 2 + h], p);
        }
        pbuf[e * 2 + h] = p;
    }
}

// comb init: cols 0..63 = layer input features, cols 64..575 = gat bias (per relation)
__global__ void comb_init(const float* __restrict__ hin, const float* __restrict__ bias512,
                          float* __restrict__ comb, int n)
{
    int idx = blockIdx.x * blockDim.x + threadIdx.x;
    if (idx >= n * 576) return;
    int node = idx / 576, j = idx % 576;
    comb[idx] = (j < 64) ? hin[(size_t)node * 64 + j] : bias512[j - 64];
}

__global__ void scatter_self(const _Float16* __restrict__ Hf, const float* __restrict__ pself,
                             const float* __restrict__ den, float* __restrict__ comb,
                             int colOff, int n)
{
    int idx = blockIdx.x * blockDim.x + threadIdx.x;
    if (idx >= n * 128) return;
    int node = idx >> 7, j = idx & 127, h = j >> 6;
    float a = pself[node * 2 + h] / den[node * 2 + h];
    atomicAdd(&comb[(size_t)node * 576 + colOff + j], a * (float)Hf[(size_t)node * 128 + j]);
}

// one wave32 per edge; lane covers 4 of the 128 channels (8B f16 gather per lane)
__global__ void scatter_edges(const int* __restrict__ srcA, const int* __restrict__ dstA, int swap,
                              const _Float16* __restrict__ Hf, const float* __restrict__ pbuf,
                              const float* __restrict__ den, float* __restrict__ comb,
                              int colOff, int E)
{
    int t = blockIdx.x * blockDim.x + threadIdx.x;
    int e = t >> 5;
    if (e >= E) return;
    int lane = t & 31;
    float p0 = pbuf[e * 2], p1 = pbuf[e * 2 + 1];
    if (p0 == 0.f && p1 == 0.f) return;       // masked edge
    int s = srcA[e], d = dstA[e];
    if (swap) { int tt = s; s = d; d = tt; }
    float a = (lane < 16) ? (p0 / den[d * 2]) : (p1 / den[d * 2 + 1]);
    const _Float16* hp = Hf + (size_t)s * 128 + lane * 4;
    float* cp = comb + (size_t)d * 576 + colOff + lane * 4;
    #pragma unroll
    for (int j = 0; j < 4; j++) atomicAdd(&cp[j], a * (float)hp[j]);
}

// ---------- MLP / norm ----------
__global__ void act_tanh(const float* __restrict__ z1, const float* __restrict__ b1,
                         _Float16* __restrict__ th, int n)
{
    int idx = blockIdx.x * blockDim.x + threadIdx.x;
    if (idx >= n * 64) return;
    int j = idx & 63;
    th[idx] = (_Float16)tanhf(z1[idx] + b1[j]);
}

// wave32 per node: z = z2 + b2 + hin (residual), then LayerNorm; writes f32 out + f16 next-layer feat
__global__ void final_ln(const float* __restrict__ z2, const float* __restrict__ b2,
                         const float* __restrict__ hin, const float* __restrict__ g,
                         const float* __restrict__ b, float* __restrict__ hout,
                         _Float16* __restrict__ hh, int n)
{
    int node = blockIdx.x * (blockDim.x >> 5) + (threadIdx.x >> 5);
    if (node >= n) return;
    int lane = threadIdx.x & 31;
    size_t base = (size_t)node * 64;
    int j0 = lane, j1 = lane + 32;
    float z0 = z2[base + j0] + b2[j0] + hin[base + j0];
    float z1 = z2[base + j1] + b2[j1] + hin[base + j1];
    float s = z0 + z1, sq = z0 * z0 + z1 * z1;
    #pragma unroll
    for (int off = 16; off > 0; off >>= 1) {
        s  += __shfl_xor(s,  off, 32);
        sq += __shfl_xor(sq, off, 32);
    }
    float mu  = s * (1.0f / 64.0f);
    float var = sq * (1.0f / 64.0f) - mu * mu;
    float inv = rsqrtf(var + 1e-5f);
    float o0 = (z0 - mu) * inv * g[j0] + b[j0];
    float o1 = (z1 - mu) * inv * g[j1] + b[j1];
    hout[base + j0] = o0; hout[base + j1] = o1;
    hh[base + j0] = (_Float16)o0; hh[base + j1] = (_Float16)o1;
}

// ---------- host orchestration ----------
extern "C" void kernel_launch(void* const* d_in, const int* in_sizes, int n_in,
                              void* d_out, int out_size, void* d_ws, size_t ws_size,
                              hipStream_t stream)
{
    const float* x        = (const float*)d_in[0];
    const int*   eidx     = (const int*)  d_in[1];
    const float* ew       = (const float*)d_in[2];
    const float* gat_W    = (const float*)d_in[3];   // [2][4][64][128]
    const float* gat_asrc = (const float*)d_in[4];   // [2][4][2][64]
    const float* gat_adst = (const float*)d_in[5];
    const float* gat_bias = (const float*)d_in[6];   // [2][4][128]
    const float* W1       = (const float*)d_in[7];   // [2][576][64]
    const float* b1       = (const float*)d_in[8];
    const float* W2       = (const float*)d_in[9];   // [2][64][64]
    const float* b2       = (const float*)d_in[10];
    const float* lng      = (const float*)d_in[11];
    const float* lnb      = (const float*)d_in[12];

    const int N = in_sizes[0] / 64;
    const int E = in_sizes[2];
    const int* srcA = eidx;
    const int* dstA = eidx + E;

    // workspace carve-up
    char* ws = (char*)d_ws; size_t off = 0;
    auto carve = [&](size_t bytes) -> void* {
        void* p = ws + off; off += (bytes + 255) & ~(size_t)255; return p;
    };
    float*     h1     = (float*)    carve((size_t)N * 64 * 4);
    _Float16*  hh     = (_Float16*) carve((size_t)N * 64 * 2);
    _Float16*  Hf     = (_Float16*) carve((size_t)N * 128 * 2);  // f16 GAT features
    float*     es     = (float*)    carve((size_t)N * 2 * 4);
    float*     edv    = (float*)    carve((size_t)N * 2 * 4);
    unsigned*  emaxB  = (unsigned*) carve((size_t)N * 2 * 4);
    float*     emaxf  = (float*)    carve((size_t)N * 2 * 4);
    float*     pself  = (float*)    carve((size_t)N * 2 * 4);
    float*     den    = (float*)    carve((size_t)N * 2 * 4);
    float*     pbuf   = (float*)    carve((size_t)E * 2 * 4);
    float*     comb   = (float*)    carve((size_t)N * 576 * 4);
    _Float16*  combh  = (_Float16*) carve((size_t)N * 576 * 2);
    float*     z1     = (float*)    carve((size_t)N * 64 * 4);
    _Float16*  th     = (_Float16*) carve((size_t)N * 64 * 2);
    float*     z2     = (float*)    carve((size_t)N * 64 * 4);
    _Float16*  gatWt  = (_Float16*) carve((size_t)8 * 128 * 64 * 2);  // [L*R][128][64]
    _Float16*  W1t    = (_Float16*) carve((size_t)2 * 64 * 576 * 2);  // [L][64][576]
    _Float16*  W2t    = (_Float16*) carve((size_t)2 * 64 * 64 * 2);   // [L][64][64]

    const int B = 256;

    // weight prep (every call; deterministic)
    transpose_f32_f16<<<CDIV(8 * 64 * 128, B), B, 0, stream>>>(gat_W, gatWt, 64, 128, 8);
    transpose_f32_f16<<<CDIV(2 * 576 * 64, B), B, 0, stream>>>(W1, W1t, 576, 64, 2);
    transpose_f32_f16<<<CDIV(2 * 64 * 64, B), B, 0, stream>>>(W2, W2t, 64, 64, 2);
    cvt_f32_to_f16<<<CDIV(N * 64, B), B, 0, stream>>>(x, hh, N * 64);

    const int rowTiles = N / 16;   // N = 50000 -> 3125 exact

    for (int l = 0; l < 2; l++) {
        const float* hin  = (l == 0) ? x  : h1;
        float*       hout = (l == 0) ? h1 : (float*)d_out;

        comb_init<<<CDIV(N * 576, B), B, 0, stream>>>(hin, gat_bias + l * 512, comb, N);

        for (int r = 0; r < 4; r++) {
            int swap = r & 1;            // relations 1,3 use reversed edges
            int wantPos = (r < 2) ? 1 : 0;
            int colOff = 64 + r * 128;

            // h_r = hin @ W[l,r]  (WMMA row-block: A fragment register-resident, f16 out)
            gemm_wmma_rowblock<true><<<CDIV(rowTiles, 8), B, 0, stream>>>(
                hh, gatWt + (size_t)(l * 4 + r) * 128 * 64, (void*)Hf, rowTiles, 8, 64);

            esed_kernel<<<CDIV(N * 2, B), B, 0, stream>>>(
                Hf, gat_asrc + (size_t)(l * 4 + r) * 128, gat_adst + (size_t)(l * 4 + r) * 128,
                es, edv, N);
            attn_self_max<<<CDIV(N * 2, B), B, 0, stream>>>(es, edv, emaxB, N * 2);
            attn_edge_max<<<CDIV(E, B), B, 0, stream>>>(srcA, dstA, ew, swap, wantPos,
                                                        es, edv, emaxB, E);
            attn_den_init<<<CDIV(N * 2, B), B, 0, stream>>>(es, edv, emaxB, emaxf, pself, den, N * 2);
            attn_edge_p<<<CDIV(E, B), B, 0, stream>>>(srcA, dstA, ew, swap, wantPos,
                                                      es, edv, emaxf, pbuf, den, E);
            scatter_self<<<CDIV(N * 128, B), B, 0, stream>>>(Hf, pself, den, comb, colOff, N);
            scatter_edges<<<CDIV(E * 32, B), B, 0, stream>>>(srcA, dstA, swap, Hf, pbuf, den,
                                                             comb, colOff, E);
        }

        // MLP: z = tanh(comb @ W1 + b1) @ W2 + b2 ; residual + LayerNorm
        cvt_f32_to_f16<<<CDIV(N * 576, B), B, 0, stream>>>(comb, combh, N * 576);
        gemm_wmma_f16<<<CDIV(rowTiles * 4, 8), B, 0, stream>>>(
            combh, W1t + (size_t)l * 64 * 576, z1, rowTiles, 4, 576);
        act_tanh<<<CDIV(N * 64, B), B, 0, stream>>>(z1, b1 + l * 64, th, N);
        gemm_wmma_rowblock<false><<<CDIV(rowTiles, 8), B, 0, stream>>>(
            th, W2t + (size_t)l * 64 * 64, (void*)z2, rowTiles, 4, 64);
        final_ln<<<CDIV(N, 8), B, 0, stream>>>(z2, b2 + l * 64, hin, lng + l * 64, lnb + l * 64,
                                               hout, hh, N);
    }
}
